// TemporalIdentityEncoder_10771777978976
// MI455X (gfx1250) — matrix operand order repaired and checked by reading, not compiled, not run on hardware
//
#include <hip/hip_runtime.h>
#include <hip/hip_bf16.h>

typedef __attribute__((ext_vector_type(16))) _Float16 v16h;
typedef __attribute__((ext_vector_type(8)))  float    v8f;

#define H_GRU 64
#define HID   128
#define APP   512
#define TLEN  16
#define G3    192          // 3 * H_GRU
#define KX    96           // augmented GRU K: 64 h + [x0, x1, 1] + zero pad
#define WAVES 4            // waves per block
#define ROWS_PER_WG (WAVES * 16)

// f16 weight scratch layout in d_ws (halves): [0,65536) Wa | [65536,98304) Wf | [98304,106496) Wt
#define WS_WA   0
#define WS_WF   65536
#define WS_WT   98304
#define WS_TOT  106496

// ---- fast transcendentals on CDNA5 hardware units ----
#if __has_builtin(__builtin_amdgcn_tanhf)
__device__ __forceinline__ float fast_tanh(float x) { return __builtin_amdgcn_tanhf(x); }
#else
__device__ __forceinline__ float fast_tanh(float x) {
    float e = __builtin_amdgcn_exp2f(x * 2.88539008177792681f);   // v_exp_f32
    return 1.0f - 2.0f * __builtin_amdgcn_rcpf(e + 1.0f);         // v_rcp_f32
}
#endif
__device__ __forceinline__ float fast_sigmoid(float x) {
    return fmaf(0.5f, fast_tanh(0.5f * x), 0.5f);   // exact identity
}

__device__ __forceinline__ v8f wmma16(v16h a, v16h b, v8f c) {
    return __builtin_amdgcn_wmma_f32_16x16x32_f16(
        false, a, false, b, (short)0, c, false, false);
}

// A fragment (16xK, 16-bit): lane row = lane&15; element j -> K = kbase+16*(lane>>4)+j
__device__ __forceinline__ v16h ldsA(const _Float16* base, int stride, int kbase, int lane) {
    const int lo = lane & 15, hi = lane >> 4;
    v16h a;
#pragma unroll
    for (int j = 0; j < 16; ++j) {
        int k = kbase + 16 * hi + j;
        a[j] = base[lo * stride + k];
    }
    return a;
}

// B fragment (Kx16 of W^T = rows of W): lane col = nbase+(lane&15); K = kbase+16*(lane>>4)+j
__device__ __forceinline__ v16h ldsB(const _Float16* W, int ldw, int nbase, int kbase, int lane) {
    const _Float16* p = W + (nbase + (lane & 15)) * ldw + kbase + 16 * (lane >> 4);
    v16h b;
#pragma unroll
    for (int j = 0; j < 16; ++j) b[j] = p[j];
    return b;
}

// B fragment from pre-converted global f16 weights: 32 contiguous bytes, no converts.
__device__ __forceinline__ v16h glbB16(const _Float16* W, int ldw, int nbase, int kbase, int lane) {
    const _Float16* p = W + (size_t)(nbase + (lane & 15)) * ldw + kbase + 16 * (lane >> 4);
    v16h b;
#pragma unroll
    for (int j = 0; j < 16; ++j) b[j] = p[j];
    return b;
}

// Fallback: B fragment from global f32 weights with inline f16 convert.
__device__ __forceinline__ v16h glbB32(const float* W, int ldw, int nbase, int kbase, int lane) {
    const float* p = W + (size_t)(nbase + (lane & 15)) * ldw + kbase + 16 * (lane >> 4);
    v16h b;
#pragma unroll
    for (int j = 0; j < 16; ++j) b[j] = (_Float16)p[j];
    return b;
}

template<bool W16>
__device__ __forceinline__ v16h wB(const float* Wf32, const _Float16* Wh,
                                   int ldw, int nbase, int kbase, int lane) {
    if constexpr (W16) return glbB16(Wh, ldw, nbase, kbase, lane);
    else               return glbB32(Wf32, ldw, nbase, kbase, lane);
}

// ---- prepass: convert Wa/Wf/Wt to f16 once (L2-resident afterwards) ----
__global__ void cvt_weights_kernel(const float* __restrict__ Wa, const float* __restrict__ Wf,
                                   const float* __restrict__ Wt, _Float16* __restrict__ dst) {
    int i = blockIdx.x * 256 + threadIdx.x;
    if (i < WS_WF)        dst[i] = (_Float16)Wa[i];
    else if (i < WS_WT)   dst[i] = (_Float16)Wf[i - WS_WF];
    else if (i < WS_TOT)  dst[i] = (_Float16)Wt[i - WS_WT];
}

template<bool W16>
__global__ __launch_bounds__(WAVES * 32)
void tie_fused_kernel(const float* __restrict__ appearance,
                      const float* __restrict__ centroid,
                      const float* __restrict__ W_ih, const float* __restrict__ W_hh,
                      const float* __restrict__ b_ih, const float* __restrict__ b_hh,
                      const float* __restrict__ Wt,  const float* __restrict__ bt,
                      const float* __restrict__ Wa,  const float* __restrict__ ba,
                      const float* __restrict__ Wf,  const float* __restrict__ bf,
                      const _Float16* __restrict__ w16,   // f16 weight scratch (may be null)
                      float* __restrict__ out) {
    // Augmented GRU B image: [192 gates][96]: cols 0..63 = W_hh, col 64/65 = W_ih (r,z only),
    // col 66 = bias (r,z: b_ih+b_hh ; n: b_hh), cols 67..95 = 0.   36 KB
    __shared__ _Float16 sWhhX[G3 * KX];
    // n-gate x-projection B image vs the a2 chunk: [64 cols][32 K]:
    // k=0 -> W_ih_n0, k=1 -> W_ih_n1, k=2 -> b_ih_n, else 0.        4 KB
    __shared__ _Float16 sWihX[H_GRU * 32];
    __shared__ _Float16 sH[WAVES][2][16 * H_GRU];         // double-buffered GRU state;
                                                          // buffer 1 reused as 16x128 feature staging

    const _Float16* wa16 = w16 + WS_WA;
    const _Float16* wf16 = w16 + WS_WF;
    const _Float16* wt16 = w16 + WS_WT;

    const int tid  = threadIdx.x;
    const int wave = tid >> 5;
    const int lane = tid & 31;
    const int lo   = lane & 15;
    const int hi   = lane >> 4;
    const long rowBase = (long)blockIdx.x * ROWS_PER_WG + wave * 16;

    // ---- cooperative staging of the augmented GRU weight images ----
    for (int i = tid; i < G3 * KX; i += WAVES * 32) {
        int g = i / KX, k = i - g * KX;
        float v;
        if (k < H_GRU)           v = W_hh[g * H_GRU + k];
        else if (k == H_GRU)     v = (g < 2 * H_GRU) ? W_ih[g * 2 + 0] : 0.0f;
        else if (k == H_GRU + 1) v = (g < 2 * H_GRU) ? W_ih[g * 2 + 1] : 0.0f;
        else if (k == H_GRU + 2) v = (g < 2 * H_GRU) ? (b_ih[g] + b_hh[g]) : b_hh[g];
        else                     v = 0.0f;
        sWhhX[i] = (_Float16)v;
    }
    for (int i = tid; i < H_GRU * 32; i += WAVES * 32) {
        int n = i >> 5, k = i & 31;
        float v = 0.0f;
        if (k == 0)      v = W_ih[(2 * H_GRU + n) * 2 + 0];
        else if (k == 1) v = W_ih[(2 * H_GRU + n) * 2 + 1];
        else if (k == 2) v = b_ih[2 * H_GRU + n];
        sWihX[i] = (_Float16)v;
    }
    for (int i = lane; i < 16 * H_GRU; i += 32) sH[wave][0][i] = (_Float16)0.0f;
    __syncthreads();

    // GRU state kept in registers in C-layout (row = v+8*hi, col = jt*16+lo)
    float h_reg[4][8];
#pragma unroll
    for (int jt = 0; jt < 4; ++jt)
#pragma unroll
        for (int v = 0; v < 8; ++v) h_reg[jt][v] = 0.0f;

    // ================= GRU over 16 timesteps (wave-private LDS) =================
    int cur = 0;
    for (int t = 0; t < TLEN; ++t) {
        // A fragments: h (K chunks 0,1) + augmented chunk 2 carrying [x0, x1, 1]
        v16h a0 = ldsA(sH[wave][cur], H_GRU, 0,  lane);
        v16h a1 = ldsA(sH[wave][cur], H_GRU, 32, lane);

        float x0 = centroid[(rowBase + lo) * (TLEN * 2) + t * 2 + 0];
        float x1 = centroid[(rowBase + lo) * (TLEN * 2) + t * 2 + 1];
        v16h a2;
#pragma unroll
        for (int j = 0; j < 16; ++j) a2[j] = (_Float16)0.0f;   // pad is zero
        a2[0] = (hi == 0) ? (_Float16)x0 : (_Float16)0.0f;     // K-local 0
        a2[1] = (hi == 0) ? (_Float16)x1 : (_Float16)0.0f;     // K-local 1
        a2[2] = (hi == 0) ? (_Float16)1.0f : (_Float16)0.0f;   // K-local 2 (bias col)

#pragma unroll
        for (int jt = 0; jt < 4; ++jt) {            // 4 col-tiles of the 64 hidden units
            v8f cr = {}, cz = {}, cn = {}, cx = {};
            cr = wmma16(a0, ldsB(sWhhX, KX, (jt)     * 16, 0,  lane), cr);
            cr = wmma16(a1, ldsB(sWhhX, KX, (jt)     * 16, 32, lane), cr);
            cr = wmma16(a2, ldsB(sWhhX, KX, (jt)     * 16, 64, lane), cr);
            cz = wmma16(a0, ldsB(sWhhX, KX, (4 + jt) * 16, 0,  lane), cz);
            cz = wmma16(a1, ldsB(sWhhX, KX, (4 + jt) * 16, 32, lane), cz);
            cz = wmma16(a2, ldsB(sWhhX, KX, (4 + jt) * 16, 64, lane), cz);
            cn = wmma16(a0, ldsB(sWhhX, KX, (8 + jt) * 16, 0,  lane), cn);
            cn = wmma16(a1, ldsB(sWhhX, KX, (8 + jt) * 16, 32, lane), cn);
            cn = wmma16(a2, ldsB(sWhhX, KX, (8 + jt) * 16, 64, lane), cn);
            cx = wmma16(a2, ldsB(sWihX, 32, jt * 16, 0, lane), cx);   // xn = x0*w0+x1*w1+b_ih_n

#pragma unroll
            for (int v = 0; v < 8; ++v) {
                const int rloc = v + 8 * hi;
                float rg = fast_sigmoid(cr[v]);                 // full arg already in acc
                float zg = fast_sigmoid(cz[v]);
                float ng = fast_tanh(fmaf(rg, cn[v], cx[v]));   // cn = hn + b_hh_n
                float hnew = fmaf(zg, h_reg[jt][v] - ng, ng);   // (1-z)*n + z*h (h from regs)
                h_reg[jt][v] = hnew;
                sH[wave][cur ^ 1][rloc * H_GRU + jt * 16 + lo] = (_Float16)hnew;
            }
        }
        cur ^= 1;
        __builtin_amdgcn_wave_barrier();   // wave-private LDS; ds ops are in-order per wave
    }
    // final hidden state in sH[wave][0]; sH[wave][1] (16x128 f16) now free -> feature staging
    _Float16* featBuf = sH[wave][1];

    // ================= appear_proj: [16x512] x [512x128] =================
    v8f acc[8];
#pragma unroll
    for (int nt = 0; nt < 8; ++nt) {
        float bv = ba[nt * 16 + lo];
#pragma unroll
        for (int v = 0; v < 8; ++v) acc[nt][v] = bv;
    }
    const float* aprow = appearance + (rowBase + lo) * APP;
    for (int kc = 0; kc < APP / 32; ++kc) {
        v16h a;
        const int kb = kc * 32 + 16 * hi;
#pragma unroll
        for (int j = 0; j < 8; ++j) {
            a[j]     = (_Float16)aprow[kb + j];
            a[j + 8] = (_Float16)aprow[kb + 8 + j];
        }
#pragma unroll
        for (int nt = 0; nt < 8; ++nt)
            acc[nt] = wmma16(a, wB<W16>(Wa, wa16, APP, nt * 16, kc * 32, lane), acc[nt]);
    }
#pragma unroll
    for (int nt = 0; nt < 8; ++nt)
#pragma unroll
        for (int v = 0; v < 8; ++v)
            featBuf[(v + 8 * hi) * HID + nt * 16 + lo] = (_Float16)acc[nt][v];
    __builtin_amdgcn_wave_barrier();

    // ================= fuse pass 1: Wf[:, 0:128] * appear_feat =================
    v8f fac[8];
#pragma unroll
    for (int nt = 0; nt < 8; ++nt) {
        float bv = bf[nt * 16 + lo];
#pragma unroll
        for (int v = 0; v < 8; ++v) fac[nt][v] = bv;
    }
#pragma unroll
    for (int kc = 0; kc < 4; ++kc) {
        v16h a = ldsA(featBuf, HID, kc * 32, lane);
#pragma unroll
        for (int nt = 0; nt < 8; ++nt)
            fac[nt] = wmma16(a, wB<W16>(Wf, wf16, 2 * HID, nt * 16, kc * 32, lane), fac[nt]);
    }
    __builtin_amdgcn_wave_barrier();

    // ================= traj_proj: [16x64] x [64x128] =================
    {
        v16h h0 = ldsA(sH[wave][0], H_GRU, 0,  lane);
        v16h h1 = ldsA(sH[wave][0], H_GRU, 32, lane);
        v8f tacc[8];
#pragma unroll
        for (int nt = 0; nt < 8; ++nt) {
            float bv = bt[nt * 16 + lo];
#pragma unroll
            for (int v = 0; v < 8; ++v) tacc[nt][v] = bv;
            tacc[nt] = wmma16(h0, wB<W16>(Wt, wt16, H_GRU, nt * 16, 0,  lane), tacc[nt]);
            tacc[nt] = wmma16(h1, wB<W16>(Wt, wt16, H_GRU, nt * 16, 32, lane), tacc[nt]);
        }
#pragma unroll
        for (int nt = 0; nt < 8; ++nt)
#pragma unroll
            for (int v = 0; v < 8; ++v)
                featBuf[(v + 8 * hi) * HID + nt * 16 + lo] = (_Float16)tacc[nt][v];
    }
    __builtin_amdgcn_wave_barrier();

    // ================= fuse pass 2: Wf[:, 128:256] * traj_feat =================
#pragma unroll
    for (int kc = 0; kc < 4; ++kc) {
        v16h a = ldsA(featBuf, HID, kc * 32, lane);
#pragma unroll
        for (int nt = 0; nt < 8; ++nt)
            fac[nt] = wmma16(a, wB<W16>(Wf, wf16, 2 * HID, nt * 16, HID + kc * 32, lane), fac[nt]);
    }

    // ================= exact GELU + store =================
#pragma unroll
    for (int nt = 0; nt < 8; ++nt) {
#pragma unroll
        for (int v = 0; v < 8; ++v) {
            float x = fac[nt][v];
            float g = 0.5f * x * (1.0f + erff(x * 0.70710678118654752f));
            out[(rowBase + v + 8 * hi) * HID + nt * 16 + lo] = g;
        }
    }
}

extern "C" void kernel_launch(void* const* d_in, const int* in_sizes, int n_in,
                              void* d_out, int out_size, void* d_ws, size_t ws_size,
                              hipStream_t stream) {
    const float* appearance = (const float*)d_in[0];
    const float* centroid   = (const float*)d_in[1];
    const float* W_ih = (const float*)d_in[2];
    const float* W_hh = (const float*)d_in[3];
    const float* b_ih = (const float*)d_in[4];
    const float* b_hh = (const float*)d_in[5];
    const float* Wt   = (const float*)d_in[6];
    const float* bt   = (const float*)d_in[7];
    const float* Wa   = (const float*)d_in[8];
    const float* ba   = (const float*)d_in[9];
    const float* Wf   = (const float*)d_in[10];
    const float* bf   = (const float*)d_in[11];
    float* out = (float*)d_out;

    const int B = in_sizes[0] / APP;          // 131072
    const int blocks = B / ROWS_PER_WG;       // 2048

    const bool use16 = ws_size >= (size_t)WS_TOT * sizeof(_Float16);
    if (use16) {
        _Float16* w16 = (_Float16*)d_ws;
        cvt_weights_kernel<<<dim3((WS_TOT + 255) / 256), dim3(256), 0, stream>>>(Wa, Wf, Wt, w16);
        tie_fused_kernel<true><<<dim3(blocks), dim3(WAVES * 32), 0, stream>>>(
            appearance, centroid, W_ih, W_hh, b_ih, b_hh,
            Wt, bt, Wa, ba, Wf, bf, w16, out);
    } else {
        tie_fused_kernel<false><<<dim3(blocks), dim3(WAVES * 32), 0, stream>>>(
            appearance, centroid, W_ih, W_hh, b_ih, b_hh,
            Wt, bt, Wa, ba, Wf, bf, (const _Float16*)d_ws, out);
    }
}
